// CalculateAttention_17239998726826
// MI455X (gfx1250) — compile-verified, matrix operand order checked by reading, not run on hardware
//
#include <hip/hip_runtime.h>

// Flash-attention for B=8, H=16, S=1024, D=64 on gfx1250 (MI455X).
// V_WMMA_F32_16X16X32_F16 with f32 accumulation, online softmax (exp2 domain),
// KV tile = 64, LDS-staged K / V^T f16 tiles, wave-private LDS P-transpose,
// async double-buffered f32 staging via GLOBAL_LOAD_ASYNC_TO_LDS_B128,
// ds_swizzle row reductions, uniform-branch mask fast path,
// register-transposed V staging (b64 LDS stores). wave32.

typedef __attribute__((ext_vector_type(16))) _Float16 v16h;
typedef __attribute__((ext_vector_type(4)))  _Float16 v4h;
typedef __attribute__((ext_vector_type(8)))  float    v8f;
typedef __attribute__((ext_vector_type(4)))  int      v4i;

#define B_ 8
#define H_ 16
#define S_ 1024
#define D_ 64
#define KV_TILE 64
#define QTILE 64          // queries per workgroup: 4 waves x 16 rows
#define NWAVES 4

#if __has_builtin(__builtin_amdgcn_global_load_async_to_lds_b128)
#define ASYNC_OK 1
#else
#define ASYNC_OK 0
#endif

typedef __attribute__((address_space(1))) v4i* gv4i_p;   // global int4*
typedef __attribute__((address_space(3))) v4i* lv4i_p;   // LDS int4*

union Op16 { v16h h; float4 f[2]; };

#define WMMA(A, Bm, C, RA) \
  __builtin_amdgcn_wmma_f32_16x16x32_f16(false, (A), false, (Bm), (short)0, (C), (RA), false)

// xor-shuffle within each 16-lane half via ds_swizzle (immediate pattern:
// offset = {xor_mask << 10 | or_mask << 5 | and_mask}, and_mask = 0x1f).
__device__ __forceinline__ float rmax16(float v) {
  v = fmaxf(v, __int_as_float(__builtin_amdgcn_ds_swizzle(__float_as_int(v), 0x041f)));
  v = fmaxf(v, __int_as_float(__builtin_amdgcn_ds_swizzle(__float_as_int(v), 0x081f)));
  v = fmaxf(v, __int_as_float(__builtin_amdgcn_ds_swizzle(__float_as_int(v), 0x101f)));
  v = fmaxf(v, __int_as_float(__builtin_amdgcn_ds_swizzle(__float_as_int(v), 0x201f)));
  return v;
}
__device__ __forceinline__ float rsum16(float v) {
  v += __int_as_float(__builtin_amdgcn_ds_swizzle(__float_as_int(v), 0x041f));
  v += __int_as_float(__builtin_amdgcn_ds_swizzle(__float_as_int(v), 0x081f));
  v += __int_as_float(__builtin_amdgcn_ds_swizzle(__float_as_int(v), 0x101f));
  v += __int_as_float(__builtin_amdgcn_ds_swizzle(__float_as_int(v), 0x201f));
  return v;
}

__device__ __forceinline__ void wait_async0() {
#if ASYNC_OK
#if __has_builtin(__builtin_amdgcn_s_wait_asynccnt)
  __builtin_amdgcn_s_wait_asynccnt(0);
#else
  asm volatile("s_wait_asynccnt 0" ::: "memory");
#endif
#endif
}

#if ASYNC_OK
// Async-copy one 64x64 f32 tile pair (K and V) into LDS staging buffers.
__device__ __forceinline__ void stage_async(const float* __restrict__ Kg,
                                            const float* __restrict__ Vg,
                                            float* dK, float* dV, int kv0, int tid) {
#pragma unroll
  for (int i = 0; i < 8; ++i) {
    int off = (tid + i * 128) * 4;          // float offset within 64x64 tile
    int row = off >> 6, col = off & 63;
    __builtin_amdgcn_global_load_async_to_lds_b128(
        (gv4i_p)(Kg + (size_t)(kv0 + row) * D_ + col),
        (lv4i_p)(dK + row * D_ + col), 0, 0);
    __builtin_amdgcn_global_load_async_to_lds_b128(
        (gv4i_p)(Vg + (size_t)(kv0 + row) * D_ + col),
        (lv4i_p)(dV + row * D_ + col), 0, 0);
  }
}
#endif

__global__ __launch_bounds__(128)
void fa_wmma_kernel(const float* __restrict__ Q, const float* __restrict__ K,
                    const float* __restrict__ V, const unsigned char* __restrict__ mask,
                    float* __restrict__ Out) {
  __shared__ _Float16 sK[KV_TILE * D_];          // [kv][d]  f16, row-major
  __shared__ _Float16 sVt[D_ * KV_TILE];         // [d][kv]  f16 (V transposed)
  __shared__ _Float16 sP[NWAVES][16 * KV_TILE];  // per-wave P tile [q][kv]
#if ASYNC_OK
  __shared__ float stF[2][2][KV_TILE * D_];      // double-buffered raw f32 K/V tiles
#endif

  const int tid  = threadIdx.x;
  const int lane = tid & 31;
  const int wave = tid >> 5;
  const int c16  = lane & 15;   // row/column within a 16-wide tile
  const int hi   = lane >> 4;
  const int base = hi * 8;      // 16-bit WMMA operand K-pattern base
  const int rofs = hi * 8;      // accumulator row offset for this half

  const int qt = blockIdx.x & 15;      // S_/QTILE == 16 query tiles per (b,h)
  const int bh = blockIdx.x >> 4;      // 0..127
  const int b  = bh >> 4;              // bh / H_
  const bool masked = mask[b] != 0;

  const size_t bhSD = (size_t)bh * S_ * D_;
  const float* Qg = Q + bhSD;
  const float* Kg = K + bhSD;
  const float* Vg = V + bhSD;
  float*       Og = Out + bhSD;

  const int q0 = qt * QTILE + wave * 16;

  // softmax in exp2 domain: fold 1/sqrt(D) * log2(e) into Q.
  const float LOG2E = 1.44269504088896f;

  // ---- Load this wave's 16x64 Q tile into A-operand layout (2 d-slices) ----
  Op16 aq0, aq1;
  {
    const float* qrow = Qg + (size_t)(q0 + c16) * D_;
    const float scale = 0.125f * LOG2E;
#pragma unroll
    for (int i = 0; i < 16; ++i) {
      int k = base + ((i >> 3) << 4) + (i & 7);
      aq0.h[i] = (_Float16)(qrow[k] * scale);
      aq1.h[i] = (_Float16)(qrow[32 + k] * scale);
    }
  }

  v8f o0 = {}, o1 = {}, o2 = {}, o3 = {};   // O accum, 4 x (16q x 16d)
  float m[8], l[8];
#pragma unroll
  for (int j = 0; j < 8; ++j) { m[j] = -3.0e38f; l[j] = 0.0f; }

  _Float16* sPw = sP[wave];
  if (masked) {
    // softmax of all-equal scores == uniform weights: P == 1, normalizer == S.
#pragma unroll
    for (int j = 0; j < 8; ++j) {
      _Float16* pr = sPw + (j + rofs) * KV_TILE;
      pr[c16]      = (_Float16)1.0f;
      pr[16 + c16] = (_Float16)1.0f;
      pr[32 + c16] = (_Float16)1.0f;
      pr[48 + c16] = (_Float16)1.0f;
    }
  }

#if ASYNC_OK
  stage_async(Kg, Vg, stF[0][0], stF[0][1], 0, tid);   // prime tile 0
#endif

  for (int kv0 = 0; kv0 < S_; kv0 += KV_TILE) {
#if ASYNC_OK
    const int buf = (kv0 >> 6) & 1;
    wait_async0();          // this wave's async copies for tile `buf` done
    __syncthreads();        // all waves done copying AND done reading f16 tiles
    if (kv0 + KV_TILE < S_)
      stage_async(Kg, Vg, stF[buf ^ 1][0], stF[buf ^ 1][1], kv0 + KV_TILE, tid);
    const float* srcK = stF[buf][0];
    const float* srcV = stF[buf][1];
#else
    __syncthreads();
    const float* srcK = Kg + (size_t)kv0 * D_;
    const float* srcV = Vg + (size_t)kv0 * D_;
#endif

    // ---- K convert pass: f32 -> f16 LDS, row-major (b128 load, b64 store) ----
#pragma unroll
    for (int i = 0; i < 8; ++i) {
      int off = (tid + i * 128) * 4;
      int row = off >> 6, col = off & 63;
      float4 kf = *(const float4*)(srcK + row * D_ + col);
      v4h hk = {(_Float16)kf.x, (_Float16)kf.y, (_Float16)kf.z, (_Float16)kf.w};
      *(v4h*)&sK[row * D_ + col] = hk;
#if !ASYNC_OK
      if (kv0 + KV_TILE < S_)
        __builtin_prefetch(srcK + (size_t)(KV_TILE + row) * D_ + col, 0, 1);
#endif
    }
    // ---- V convert/transpose: 4x4 register sub-block -> 4 b64 LDS stores ----
#pragma unroll
    for (int ci = 0; ci < 2; ++ci) {
      int linear = tid + ci * 128;             // 0..255 sub-blocks
      int row = (linear >> 4) * 4;             // kv base (4 rows)
      int col = (linear & 15) * 4;             // d  base (4 cols)
      float4 r0 = *(const float4*)(srcV + (size_t)(row + 0) * D_ + col);
      float4 r1 = *(const float4*)(srcV + (size_t)(row + 1) * D_ + col);
      float4 r2 = *(const float4*)(srcV + (size_t)(row + 2) * D_ + col);
      float4 r3 = *(const float4*)(srcV + (size_t)(row + 3) * D_ + col);
      v4h t0 = {(_Float16)r0.x, (_Float16)r1.x, (_Float16)r2.x, (_Float16)r3.x};
      v4h t1 = {(_Float16)r0.y, (_Float16)r1.y, (_Float16)r2.y, (_Float16)r3.y};
      v4h t2 = {(_Float16)r0.z, (_Float16)r1.z, (_Float16)r2.z, (_Float16)r3.z};
      v4h t3 = {(_Float16)r0.w, (_Float16)r1.w, (_Float16)r2.w, (_Float16)r3.w};
      *(v4h*)&sVt[(col + 0) * KV_TILE + row] = t0;
      *(v4h*)&sVt[(col + 1) * KV_TILE + row] = t1;
      *(v4h*)&sVt[(col + 2) * KV_TILE + row] = t2;
      *(v4h*)&sVt[(col + 3) * KV_TILE + row] = t3;
#if !ASYNC_OK
      if (kv0 + KV_TILE < S_)
        __builtin_prefetch(srcV + (size_t)(KV_TILE + row) * D_ + col, 0, 1);
#endif
    }
    __syncthreads();

    if (!masked) {
      // ---- S = Q K^T : preload all 8 B operands, then back-to-back WMMAs ----
      v8f c0 = {}, c1 = {}, c2 = {}, c3 = {};
      Op16 bks[8];
#pragma unroll
      for (int t = 0; t < 8; ++t) {          // t = s*4 + n
        int n = t & 3, s = t >> 2;
        bks[t].f[0] = *(const float4*)&sK[(n * 16 + c16) * D_ + s * 32 + base];
        bks[t].f[1] = *(const float4*)&sK[(n * 16 + c16) * D_ + s * 32 + base + 16];
      }
      c0 = WMMA(aq0.h, bks[0].h, c0, false);
      c1 = WMMA(aq0.h, bks[1].h, c1, true);
      c2 = WMMA(aq0.h, bks[2].h, c2, true);
      c3 = WMMA(aq0.h, bks[3].h, c3, true);
      c0 = WMMA(aq1.h, bks[4].h, c0, false);
      c1 = WMMA(aq1.h, bks[5].h, c1, true);
      c2 = WMMA(aq1.h, bks[6].h, c2, true);
      c3 = WMMA(aq1.h, bks[7].h, c3, true);

      // ---- online softmax (exp2 domain); rows j, j+8 in lane halves ----
#pragma unroll
      for (int j = 0; j < 8; ++j) {
        float s0 = c0[j], s1 = c1[j], s2 = c2[j], s3 = c3[j];
        float rmax  = rmax16(fmaxf(fmaxf(s0, s1), fmaxf(s2, s3)));
        float mnew  = fmaxf(m[j], rmax);
        float alpha = exp2f(m[j] - mnew);        // first iter: exp2(-inf)=0
        float p0 = exp2f(s0 - mnew);
        float p1 = exp2f(s1 - mnew);
        float p2 = exp2f(s2 - mnew);
        float p3 = exp2f(s3 - mnew);
        l[j] = l[j] * alpha + rsum16((p0 + p1) + (p2 + p3));
        m[j] = mnew;
        o0[j] *= alpha; o1[j] *= alpha; o2[j] *= alpha; o3[j] *= alpha;
        _Float16* pr = sPw + (j + rofs) * KV_TILE;  // transpose P through LDS
        pr[c16]      = (_Float16)p0;
        pr[16 + c16] = (_Float16)p1;
        pr[32 + c16] = (_Float16)p2;
        pr[48 + c16] = (_Float16)p3;
      }
    }

    // ---- O += P x V : preload 8 V operands, then back-to-back WMMAs ----
    {
      Op16 bvs[8];
#pragma unroll
      for (int t = 0; t < 8; ++t) {          // t = s*4 + n
        int n = t & 3, s = t >> 2;
        bvs[t].f[0] = *(const float4*)&sVt[(n * 16 + c16) * KV_TILE + s * 32 + base];
        bvs[t].f[1] = *(const float4*)&sVt[(n * 16 + c16) * KV_TILE + s * 32 + base + 16];
      }
      Op16 ap0, ap1;
      ap0.f[0] = *(const float4*)&sPw[c16 * KV_TILE + base];
      ap0.f[1] = *(const float4*)&sPw[c16 * KV_TILE + base + 16];
      ap1.f[0] = *(const float4*)&sPw[c16 * KV_TILE + 32 + base];
      ap1.f[1] = *(const float4*)&sPw[c16 * KV_TILE + 32 + base + 16];
      o0 = WMMA(ap0.h, bvs[0].h, o0, false);
      o1 = WMMA(ap0.h, bvs[1].h, o1, true);
      o2 = WMMA(ap0.h, bvs[2].h, o2, true);
      o3 = WMMA(ap0.h, bvs[3].h, o3, true);
      o0 = WMMA(ap1.h, bvs[4].h, o0, false);
      o1 = WMMA(ap1.h, bvs[5].h, o1, true);
      o2 = WMMA(ap1.h, bvs[6].h, o2, true);
      o3 = WMMA(ap1.h, bvs[7].h, o3, true);
    }
  }

  // ---- normalize and store: lanes 0-15 = d columns, rows in VGPR index ----
#pragma unroll
  for (int j = 0; j < 8; ++j) {
    float rl = masked ? (1.0f / (float)S_) : (1.0f / l[j]);
    int q = q0 + j + rofs;
    float* orow = Og + (size_t)q * D_ + c16;
    orow[0]  = o0[j] * rl;
    orow[16] = o1[j] * rl;
    orow[32] = o2[j] * rl;
    orow[48] = o3[j] * rl;
  }
}

extern "C" void kernel_launch(void* const* d_in, const int* in_sizes, int n_in,
                              void* d_out, int out_size, void* d_ws, size_t ws_size,
                              hipStream_t stream) {
  (void)in_sizes; (void)n_in; (void)out_size; (void)d_ws; (void)ws_size;
  const float* Q = (const float*)d_in[0];
  const float* K = (const float*)d_in[1];
  const float* V = (const float*)d_in[2];
  const unsigned char* mask = (const unsigned char*)d_in[3];  // jnp.bool_ = 1B/elem
  float* out = (float*)d_out;

  dim3 grid(B_ * H_ * (S_ / QTILE));   // 2048 workgroups
  dim3 block(128);                     // 4 wave32 waves
  hipLaunchKernelGGL(fa_wmma_kernel, grid, block, 0, stream, Q, K, V, mask, out);
}
